// probreweighting_87651692577007
// MI455X (gfx1250) — compile-verified, matrix operand order checked by reading, not run on hardware
//
#include <hip/hip_runtime.h>
#include <hip/hip_bf16.h>

// ---------------------------------------------------------------------------
// probreweighting for MI455X (gfx1250, wave32)
//
// Phase 2 (302 MB stream @ 23.3 TB/s ~= 13 us) dominates -> non-temporal
// b128 load/mul/store. Phase 1 histogram is done with
// V_WMMA_I32_16X16X64_IU8: A = all-ones, B = one-hot(label==class) bytes,
// D[m][n] = count(label==n) over 64 labels per issue, no LDS-atomic
// contention. A being uniform 1s makes the A VGPR layout irrelevant; B only
// needs "one column per lane, K split across half-wave groups", matching the
// ISA 8-bit B layout.
// ---------------------------------------------------------------------------

typedef int   v8i  __attribute__((ext_vector_type(8)));
typedef int   iv4  __attribute__((ext_vector_type(4)));
typedef float fv4  __attribute__((ext_vector_type(4)));

#define NUM_CLASSES 9
#define BATCH 16
#define HW (512 * 512)               // 262144 labels / pixels per plane
#define HIST_WGS_PER_BATCH 32
#define HIST_BLOCK 256               // 8 waves
#define CHUNKS_PER_WAVE 16           // 8 waves * 32 wgs * 16 chunks * 64 = 262144

__global__ void zero_hist_kernel(int* __restrict__ hist) {
    if (threadIdx.x < BATCH * NUM_CLASSES) hist[threadIdx.x] = 0;
}

// grid (16, 32), block 256. Each wave bins 1024 labels via 16 chained WMMAs.
__global__ void hist_wmma_kernel(const int* __restrict__ labels,
                                 int* __restrict__ hist) {
    const int b    = blockIdx.x;
    const int wave = threadIdx.x >> 5;
    const int lane = threadIdx.x & 31;
    const int col  = lane & 15;      // B-matrix column this lane owns
    const int half = lane >> 4;      // which 32-label half of the 64-chunk

    const int waveBase = b * HW + (blockIdx.y * 8 + wave) * (CHUNKS_PER_WAVE * 64);

    v8i aOnes;
#pragma unroll
    for (int i = 0; i < 8; ++i) aOnes[i] = 0x01010101;   // A[m][k] = 1 for all
    v8i acc = {};

    const int* p = labels + waveBase + half * 32;
    for (int c = 0; c < CHUNKS_PER_WAVE; ++c) {
        v8i bm;
#pragma unroll
        for (int v = 0; v < 8; ++v) {
            iv4 l = *(const iv4*)(p + v * 4);            // 16 lanes broadcast-read
            unsigned pk = (unsigned)(l.x == col)
                        | ((unsigned)(l.y == col) << 8)
                        | ((unsigned)(l.z == col) << 16)
                        | ((unsigned)(l.w == col) << 24);
            bm[v] = (int)pk;
        }
        // D = A(1s) x B(one-hot) + C  ->  per-column label counts
        acc = __builtin_amdgcn_wmma_i32_16x16x64_iu8(
            /*sgn_a=*/false, aOnes, /*sgn_b=*/false, bm, acc,
            /*reuse_a=*/false, /*reuse_b=*/false);
        p += 64;
    }

    // All D rows are identical; lane n (n<9) holds count for class n in acc[0].
    if (lane < NUM_CLASSES) atomicAdd(&hist[b * NUM_CLASSES + lane], acc[0]);
}

// One thread per batch sample: exact reference math.
__global__ void weights_kernel(const int* __restrict__ hist,
                               float* __restrict__ w) {
    const int b = threadIdx.x;
    if (b >= BATCH) return;

    float h[NUM_CLASSES];
    float total = 0.f;
#pragma unroll
    for (int c = 0; c < NUM_CLASSES; ++c) {
        h[c] = (float)hist[b * NUM_CLASSES + c];
        total += h[c];
    }
    float logh[NUM_CLASSES];
    float cnt = 0.f, sum = 0.f;
#pragma unroll
    for (int c = 0; c < NUM_CLASSES; ++c) {
        float pr = h[c] / total;
        logh[c] = (h[c] > 0.f) ? -logf(pr) : 0.f;
        if (h[c] > 0.f) { cnt += 1.f; sum += logh[c]; }
    }
    const float mean = sum / cnt;
    float var = 0.f;
#pragma unroll
    for (int c = 0; c < NUM_CLASSES; ++c) {
        if (h[c] > 0.f) { float d = logh[c] - mean; var += d * d; }
    }
    var /= cnt;                       // population (ddof=0)
    const float sd = sqrtf(var);
#pragma unroll
    for (int c = 0; c < NUM_CLASSES; ++c) {
        w[b * NUM_CLASSES + c] =
            (logh[c] != 0.f) ? ((logh[c] - mean) / sd * 0.1f + 1.0f) : 1.0f;
    }
}

// 256 blocks per (b,c) plane -> per-block-uniform (scalar) weight load.
// One NT float4 per thread: pure streaming b128 traffic, L2 left for labels.
__global__ void scale_kernel(const fv4* __restrict__ in,
                             const float* __restrict__ w,
                             fv4* __restrict__ out) {
    const unsigned plane = blockIdx.x >> 8;               // 65536/256 f4s per plane
    const float wv = w[plane];
    const unsigned i = blockIdx.x * 256u + threadIdx.x;
    fv4 v = __builtin_nontemporal_load(in + i);
    v = v * wv;
    __builtin_nontemporal_store(v, out + i);
}

extern "C" void kernel_launch(void* const* d_in, const int* in_sizes, int n_in,
                              void* d_out, int out_size, void* d_ws, size_t ws_size,
                              hipStream_t stream) {
    const float* preds  = (const float*)d_in[0];
    const int*   labels = (const int*)d_in[1];
    float*       out    = (float*)d_out;

    int*   hist = (int*)d_ws;                          // 144 ints
    float* wts  = (float*)((char*)d_ws + 1024);        // 144 floats

    zero_hist_kernel<<<1, 256, 0, stream>>>(hist);

    dim3 hgrid(BATCH, HIST_WGS_PER_BATCH, 1);
    hist_wmma_kernel<<<hgrid, HIST_BLOCK, 0, stream>>>(labels, hist);

    weights_kernel<<<1, 32, 0, stream>>>(hist, wts);

    const unsigned nF4    = (unsigned)(BATCH * NUM_CLASSES * HW) / 4u; // 9,437,184
    const unsigned blocks = nF4 / 256u;                                // 36,864
    scale_kernel<<<blocks, 256, 0, stream>>>((const fv4*)preds, wts, (fv4*)out);
}
